// BipartiteGNNConvVariableToFactor_3418793967970
// MI455X (gfx1250) — matrix operand order repaired and checked by reading, not compiled, not run on hardware
//
#include <hip/hip_runtime.h>

typedef __attribute__((ext_vector_type(16))) _Float16 v16h;
typedef __attribute__((ext_vector_type(8)))  float    v8f;

#define D      128
#define TWO_D  256
#define STRIDE_A 264   // halves per LDS row (528B row -> lane m hits bank 4m, conflict-free)

union AFrag { v16h v; unsigned u[8]; _Float16 h[16]; };
union H4    { _Float16 h[4]; uint2 u2; };

// ---- B fragment: W row-major [256][128] f32 -> v16h covering K in [kt*32, kt*32+32), column n
__device__ inline v16h load_b_frag_f32(const float* __restrict__ W, int kt, int n, int lane) {
    AFrag b;
    const int kbase = kt * 32 + ((lane & 16) ? 8 : 0);
#pragma unroll
    for (int v = 0; v < 8; ++v) {
        float f0 = W[(kbase + ((v < 4) ? (2 * v) : (8 + 2 * v))    ) * D + n];
        float f1 = W[(kbase + ((v < 4) ? (2 * v) : (8 + 2 * v)) + 1) * D + n];
        b.h[2 * v]     = (_Float16)f0;
        b.h[2 * v + 1] = (_Float16)f1;
    }
    return b.v;
}

// ---- A fragment from LDS-staged f16 tile (16 rows x 256 K, padded stride)
__device__ inline v16h load_a_frag_lds(const _Float16* __restrict__ sA, int kt, int lane) {
    AFrag a;
    const int m    = lane & 15;
    const int base = (lane & 16) ? 8 : 0;
#pragma unroll
    for (int v = 0; v < 8; ++v) {
        const int c = ((v < 4) ? (2 * v) : (8 + 2 * v)) + base + kt * 32;  // even -> 4B aligned
        a.u[v] = *(const unsigned*)&sA[m * STRIDE_A + c];
    }
    return a.v;
}

// convert+store one 64B row segment (4 float4) into LDS as f16
__device__ inline void stage_seg(const float* __restrict__ srcRow, _Float16* __restrict__ dst,
                                 int kbase) {
#pragma unroll
    for (int i = 0; i < 4; ++i) {
        const float4 f = *(const float4*)&srcRow[kbase + i * 4];
        H4 t;
        t.h[0] = (_Float16)f.x; t.h[1] = (_Float16)f.y;
        t.h[2] = (_Float16)f.z; t.h[3] = (_Float16)f.w;
        *(uint2*)&dst[i * 4] = t.u2;
    }
}

__global__ void zero_kernel(float* __restrict__ p, long n) {
    long i = (long)blockIdx.x * blockDim.x + threadIdx.x;
    long stride = (long)gridDim.x * blockDim.x;
    for (long j = i * 4; j < n; j += stride * 4) {
        if (j + 3 < n) {
            *(float4*)&p[j] = make_float4(0.f, 0.f, 0.f, 0.f);
        } else {
            for (long k = j; k < n; ++k) p[k] = 0.f;
        }
    }
}

// ---- Kernel 1: per-edge message GEMM + scatter-add into aggr (== d_out)
__global__ void __launch_bounds__(256)
edge_msg_kernel(const float* __restrict__ vars,
                const float* __restrict__ facs,
                const int*   __restrict__ senders,
                const int*   __restrict__ receivers,
                const float* __restrict__ W_msg,
                const float* __restrict__ b_msg,
                float*       __restrict__ aggr,
                int n_edges, int n_tiles) {
    __shared__ __align__(16) _Float16 sA[16 * STRIDE_A];
    __shared__ __align__(16) int      sRecv[16];

    const int t    = (int)threadIdx.x;
    const int lane = t & 31;
    const int wave = t >> 5;                    // 0..7, owns 16 output columns
    const int n    = wave * 16 + (lane & 15);   // global output column 0..127

    // staging role: each thread owns one 64B segment of one row
    const int srow = t >> 4;                    // 0..15 (edge row within tile)
    const int sseg = t & 15;                    // 0..15 (which 16-float segment)
    const int skb  = (sseg & 7) * 16;           // float offset within the 128-float half
    const bool from_facs = (sseg < 8);
    _Float16* sdst = &sA[srow * STRIDE_A + (from_facs ? 0 : D) + skb];

    // W_msg fragments resident in VGPRs for the whole loop (8 kt * 8 VGPRs)
    v16h Bfrag[8];
#pragma unroll
    for (int kt = 0; kt < 8; ++kt) Bfrag[kt] = load_b_frag_f32(W_msg, kt, n, lane);
    const float bias = b_msg[n];

    for (int tile = blockIdx.x; tile < n_tiles; tile += gridDim.x) {
        const int e0 = tile * 16;

        // prefetch next tile's edge indices while we work on this one
        const int tile_next = tile + gridDim.x;
        if (tile_next < n_tiles && t < 16) {
            __builtin_prefetch(&receivers[tile_next * 16 + t], 0, 0);
            __builtin_prefetch(&senders[tile_next * 16 + t], 0, 0);
        }

        // Stage A tile: one index load + one 64B contiguous gather per thread
        {
            int e = e0 + srow; if (e >= n_edges) e = n_edges - 1;
            const long r = from_facs ? (long)receivers[e] : (long)senders[e];
            const float* srcRow = from_facs ? &facs[r * D] : &vars[r * D];
            stage_seg(srcRow, sdst, skb);
        }
        if (t < 16) {
            int e = e0 + t; if (e >= n_edges) e = n_edges - 1;
            sRecv[t] = receivers[e];
        }
        __syncthreads();

        v8f acc = {};
#pragma unroll
        for (int kt = 0; kt < 8; ++kt) {
            v16h a = load_a_frag_lds(sA, kt, lane);
            acc = __builtin_amdgcn_wmma_f32_16x16x32_f16(
                false, a, false, Bfrag[kt], (short)0, acc, false, false);
        }

        // bias + relu + scatter-add; receiver rows from LDS (two 16B ds loads)
        const int mbase = (lane & 16) ? 8 : 0;
        int ridx[8];
        *(int4*)&ridx[0] = *(const int4*)&sRecv[mbase];
        *(int4*)&ridx[4] = *(const int4*)&sRecv[mbase + 4];

        if (e0 + 16 <= n_edges) {               // uniform fast path: no per-row masking
#pragma unroll
            for (int v = 0; v < 8; ++v) {
                float val = acc[v] + bias;
                val = val > 0.f ? val : 0.f;
                atomicAdd(&aggr[(long)ridx[v] * D + n], val);
            }
        } else {                                // guarded tail (cold)
#pragma unroll
            for (int v = 0; v < 8; ++v) {
                if (e0 + mbase + v < n_edges) {
                    float val = acc[v] + bias;
                    val = val > 0.f ? val : 0.f;
                    atomicAdd(&aggr[(long)ridx[v] * D + n], val);
                }
            }
        }
        __syncthreads();
    }
}

// ---- Kernel 2: combine MLP out = relu([factors | aggr] @ W_comb + b); aggr aliased with out
__global__ void __launch_bounds__(256)
combine_kernel(const float* __restrict__ facs,
               const float* __restrict__ W_comb,
               const float* __restrict__ b_comb,
               float*       __restrict__ out,     // input rows (aggr) and output rows
               int n_factors, int n_tiles) {
    __shared__ __align__(16) _Float16 sA[16 * STRIDE_A];

    const int t    = (int)threadIdx.x;
    const int lane = t & 31;
    const int wave = t >> 5;
    const int n    = wave * 16 + (lane & 15);

    const int srow = t >> 4;
    const int sseg = t & 15;
    const int skb  = (sseg & 7) * 16;
    const bool from_facs = (sseg < 8);
    _Float16* sdst = &sA[srow * STRIDE_A + (from_facs ? 0 : D) + skb];

    v16h Bfrag[8];
#pragma unroll
    for (int kt = 0; kt < 8; ++kt) Bfrag[kt] = load_b_frag_f32(W_comb, kt, n, lane);
    const float bias = b_comb[n];

    for (int tile = blockIdx.x; tile < n_tiles; tile += gridDim.x) {
        const int f0 = tile * 16;

        {
            int f = f0 + srow; if (f >= n_factors) f = n_factors - 1;
            const float* srcRow = from_facs ? &facs[(long)f * D] : &out[(long)f * D];
            stage_seg(srcRow, sdst, skb);
        }
        __syncthreads();   // all reads of out-rows complete before any write below

        v8f acc = {};
#pragma unroll
        for (int kt = 0; kt < 8; ++kt) {
            v16h a = load_a_frag_lds(sA, kt, lane);
            acc = __builtin_amdgcn_wmma_f32_16x16x32_f16(
                false, a, false, Bfrag[kt], (short)0, acc, false, false);
        }

        const int mbase = (lane & 16) ? 8 : 0;
        if (f0 + 16 <= n_factors) {             // uniform fast path
#pragma unroll
            for (int v = 0; v < 8; ++v) {
                float val = acc[v] + bias;
                out[(long)(f0 + mbase + v) * D + n] = val > 0.f ? val : 0.f;
            }
        } else {
#pragma unroll
            for (int v = 0; v < 8; ++v) {
                const int f = f0 + mbase + v;
                if (f < n_factors) {
                    float val = acc[v] + bias;
                    out[(long)f * D + n] = val > 0.f ? val : 0.f;
                }
            }
        }
        __syncthreads();
    }
}

extern "C" void kernel_launch(void* const* d_in, const int* in_sizes, int n_in,
                              void* d_out, int out_size, void* d_ws, size_t ws_size,
                              hipStream_t stream) {
    (void)n_in; (void)out_size; (void)d_ws; (void)ws_size;
    const float* vars      = (const float*)d_in[0];
    const float* facs      = (const float*)d_in[1];
    const int*   senders   = (const int*)  d_in[2];
    const int*   receivers = (const int*)  d_in[3];
    const float* W_msg     = (const float*)d_in[4];
    const float* b_msg     = (const float*)d_in[5];
    const float* W_comb    = (const float*)d_in[6];
    const float* b_comb    = (const float*)d_in[7];
    float* out = (float*)d_out;

    const int n_edges   = in_sizes[2];
    const int n_factors = in_sizes[1] / D;
    const long out_elems = (long)n_factors * D;

    // 1) zero the aggregation buffer (d_out doubles as aggr)
    {
        long vec = (out_elems + 3) / 4;
        int blocks = (int)((vec + 255) / 256);
        if (blocks > 4096) blocks = 4096;
        zero_kernel<<<blocks, 256, 0, stream>>>(out, out_elems);
    }
    // 2) edge messages + scatter
    {
        int tiles = (n_edges + 15) / 16;
        int blocks = tiles < 2048 ? tiles : 2048;
        edge_msg_kernel<<<blocks, 256, 0, stream>>>(
            vars, facs, senders, receivers, W_msg, b_msg, out, n_edges, tiles);
    }
    // 3) combine MLP (in-place over aggr rows)
    {
        int tiles = (n_factors + 15) / 16;
        int blocks = tiles < 2048 ? tiles : 2048;
        combine_kernel<<<blocks, 256, 0, stream>>>(
            facs, W_comb, b_comb, out, n_factors, tiles);
    }
}
